// SelfAttentionHead_13752485282462
// MI455X (gfx1250) — matrix operand order repaired and checked by reading, not compiled, not run on hardware
//
#include <hip/hip_runtime.h>
#include <hip/hip_bf16.h>

// B=8, T=2048, E=1024, H=64. out[b,t,h] fp32.
// Inputs: x[8,2048,1024] f32, Wk[64,1024] f32, Wq[64,1024] f32, Wv[64,1024] f32, mask[8,2048] i32.

#define Bb 8
#define Tt 2048
#define Ee 1024
#define Hh 64
#define BT (Bb * Tt)

typedef __bf16 v16bf __attribute__((ext_vector_type(16)));
typedef float  v8f   __attribute__((ext_vector_type(8)));

union BF16Frag {
    uint4  u[2];
    v16bf  v;
};

__device__ __forceinline__ unsigned rne_bits(float f) {
    // fp32 bits with bf16 round-to-nearest-even applied; result bf16 = bits[31:16]
    unsigned u = __float_as_uint(f);
    return u + 0x7FFFu + ((u >> 16) & 1u);   // v_bfe_u32 + v_add3_u32
}

__device__ __forceinline__ unsigned short f2bf(float f) {
    return (unsigned short)(rne_bits(f) >> 16);
}

// Pack two fp32 -> packed bf16x2 (low = lo, high = hi).
__device__ __forceinline__ unsigned pk_bf2(float lo, float hi) {
#if __has_builtin(__builtin_amdgcn_cvt_pk_bf16_f32)
    auto p = __builtin_amdgcn_cvt_pk_bf16_f32(lo, hi);   // v_cvt_pk_bf16_f32, RNE
    return __builtin_bit_cast(unsigned, p);
#else
    // one v_perm_b32 packs bytes [3:2] of each rounded dword: {hi[31:16], lo[31:16]}
    return __builtin_amdgcn_perm(rne_bits(hi), rne_bits(lo), 0x07060302u);
#endif
}

// A-fragment (16x32 bf16, MxK) from a bf16 matrix: lane holds row m = lane&15.
// lanes 0-15: K = {k0..k0+7, k0+16..k0+23}; lanes 16-31: K = {k0+8..k0+15, k0+24..k0+31}.
__device__ __forceinline__ v16bf load_frag_a(const unsigned short* base, int ld, int row0, int k0) {
    const int lane = threadIdx.x & 31;
    const int hi   = lane >> 4;
    const int m    = lane & 15;
    const unsigned short* p = base + (size_t)(row0 + m) * ld + k0 + hi * 8;
    BF16Frag f;
    f.u[0] = *(const uint4*)(p);        // K = k0+hi*8 .. +7
    f.u[1] = *(const uint4*)(p + 16);   // K = k0+16+hi*8 .. +7
    return f.v;
}

// A-fragment built from an fp32 matrix, converted to bf16 in-register (fused cvt).
__device__ __forceinline__ v16bf load_frag_a_f32(const float* base, int ld, int row0, int k0) {
    const int lane = threadIdx.x & 31;
    const int hi   = lane >> 4;
    const int m    = lane & 15;
    const float* p = base + (size_t)(row0 + m) * ld + k0 + hi * 8;
    float4 c0 = *(const float4*)(p);
    float4 c1 = *(const float4*)(p + 4);
    float4 c2 = *(const float4*)(p + 16);
    float4 c3 = *(const float4*)(p + 20);
    BF16Frag f;
    f.u[0].x = pk_bf2(c0.x, c0.y);
    f.u[0].y = pk_bf2(c0.z, c0.w);
    f.u[0].z = pk_bf2(c1.x, c1.y);
    f.u[0].w = pk_bf2(c1.z, c1.w);
    f.u[1].x = pk_bf2(c2.x, c2.y);
    f.u[1].y = pk_bf2(c2.z, c2.w);
    f.u[1].z = pk_bf2(c3.x, c3.y);
    f.u[1].w = pk_bf2(c3.z, c3.w);
    return f.v;
}

// B-fragment (32x16 bf16, KxN): lane holds column n0+(lane&15); K = {k0+hi*8..+7, k0+16+hi*8..+7}.
__device__ __forceinline__ v16bf load_frag_b(const unsigned short* base, int ld, int k0, int n0) {
    const int lane = threadIdx.x & 31;
    const int hi   = lane >> 4;
    const unsigned short* p = base + (size_t)(k0 + hi * 8) * ld + n0 + (lane & 15);
    BF16Frag f;
#pragma unroll
    for (int s = 0; s < 8; ++s) {
        f.v[s]     = __builtin_bit_cast(__bf16, p[(size_t)s * ld]);
        f.v[s + 8] = __builtin_bit_cast(__bf16, p[(size_t)(s + 16) * ld]);
    }
    return f.v;
}

#define WMMA_BF16(A, Bm, C) \
    __builtin_amdgcn_wmma_f32_16x16x32_bf16(false, (A), false, (Bm), (short)0, (C), false, false)

// ---------------- Kernel 1: pack Wt[e][0..191] = bf16 of [Wq|Wk|Wv] transposed ----------------
__global__ void cvt_w_kernel(const float* __restrict__ Wk, const float* __restrict__ Wq,
                             const float* __restrict__ Wv, unsigned short* __restrict__ Wt) {
    int i = blockIdx.x * blockDim.x + threadIdx.x;   // over Ee*192
    if (i >= Ee * 192) return;
    int e = i / 192, j = i - e * 192;
    const float* W = (j < 64) ? Wq : (j < 128) ? Wk : Wv;
    int h = j & 63;
    Wt[(size_t)e * 192 + j] = f2bf(W[(size_t)h * Ee + e]);
}

// ---------------- Kernel 2: projections  Q/K/V = x @ Wt (M=BT, N=192, K=1024) ----------------
// fp32 x is read ONCE (64 MB) and converted to bf16 fragments in-register
// (bfe+add3+perm RNE pack, co-executing with the XDL pipe).
// Block = 4 waves; block mt owns rows [16*mt,16*mt+16); wave w owns N-cols [48w,48w+48).
__global__ __launch_bounds__(128) void proj_kernel(const float* __restrict__ x,
                                                   const unsigned short* __restrict__ Wt,
                                                   unsigned short* __restrict__ Qb,
                                                   unsigned short* __restrict__ Ktb,
                                                   unsigned short* __restrict__ Vb) {
    const int mt   = blockIdx.x;           // 0..BT/16-1
    const int wave = threadIdx.x >> 5;     // 0..3
    const int lane = threadIdx.x & 31;
    const int hi   = lane >> 4;
    const int nlo  = lane & 15;
    const int row0 = mt * 16;

    v8f acc0 = {}, acc1 = {}, acc2 = {};
#pragma unroll 2
    for (int e0 = 0; e0 < Ee; e0 += 32) {
        v16bf a  = load_frag_a_f32(x, Ee, row0, e0);
        v16bf b0 = load_frag_b(Wt, 192, e0, wave * 48 + 0);
        v16bf b1 = load_frag_b(Wt, 192, e0, wave * 48 + 16);
        v16bf b2 = load_frag_b(Wt, 192, e0, wave * 48 + 32);
        acc0 = WMMA_BF16(a, b0, acc0);
        acc1 = WMMA_BF16(a, b1, acc1);
        acc2 = WMMA_BF16(a, b2, acc2);
    }

    // Scatter-store: the 16-col group of each (wave,j) lies entirely inside one of
    // Q/K/V (boundaries are multiples of 16), so branch on the lane-invariant cbase.
    v8f accs[3] = {acc0, acc1, acc2};
#pragma unroll
    for (int j = 0; j < 3; ++j) {
        const int cbase = wave * 48 + j * 16;   // wave-uniform, 0..176
        const int c = cbase + nlo;
        if (cbase < 64) {                        // Q, row-major [BT,64]
#pragma unroll
            for (int r = 0; r < 8; ++r) {
                const int row = row0 + r + 8 * hi;
                Qb[(size_t)row * Hh + c] = f2bf(accs[j][r]);
            }
        } else if (cbase < 128) {                // K transposed, [B][64][T]
            const int h = c - 64;
#pragma unroll
            for (int r = 0; r < 8; ++r) {
                const int row = row0 + r + 8 * hi;
                const int bidx = row >> 11, t = row & (Tt - 1);
                Ktb[((size_t)bidx * Hh + h) * Tt + t] = f2bf(accs[j][r]);
            }
        } else {                                 // V, row-major [BT,64]
            const int h = c - 128;
#pragma unroll
            for (int r = 0; r < 8; ++r) {
                const int row = row0 + r + 8 * hi;
                Vb[(size_t)row * Hh + h] = f2bf(accs[j][r]);
            }
        }
    }
}

// ---------------- Kernel 3: flash attention, one wave per 16-row q-tile ----------------
__global__ __launch_bounds__(32) void attn_kernel(const unsigned short* __restrict__ Qb,
                                                  const unsigned short* __restrict__ Ktb,
                                                  const unsigned short* __restrict__ Vb,
                                                  const int* __restrict__ am,
                                                  float* __restrict__ out) {
    __shared__ __align__(16) unsigned short Pls[16 * 32];   // P tile bf16, row-major 16x32

    const int tile  = blockIdx.x;            // 0..B*T/16-1
    const int b     = tile >> 7;             // T/16 = 128 q-tiles per batch
    const int qbase = (tile & 127) * 16;
    const int lane  = threadIdx.x & 31;
    const int hi    = lane >> 4;
    const int nlo   = lane & 15;
    const float scale = 0.125f;              // 64^-0.5

    const unsigned short* Kt_b = Ktb + (size_t)b * Hh * Tt;
    const unsigned short* V_b  = Vb + (size_t)b * Tt * Hh;
    const int* am_b = am + b * Tt;

    // Q A-fragments for contraction over H=64 (two K=32 steps), rows qbase..qbase+15
    v16bf qf0 = load_frag_a(Qb, Hh, b * Tt + qbase, 0);
    v16bf qf1 = load_frag_a(Qb, Hh, b * Tt + qbase, 32);

    v8f o0 = {}, o1 = {}, o2 = {}, o3 = {};
    float mrow[8], lrow[8];
#pragma unroll
    for (int r = 0; r < 8; ++r) { mrow[r] = -1e30f; lrow[r] = 0.f; }

#pragma unroll 1
    for (int kbase = 0; kbase <= qbase + 15; kbase += 32) {
        // ---- prefetch next kv block's K/V cachelines (hidden under softmax/LDS phase)
        const int knext = kbase + 32;
        if (knext <= qbase + 15) {
            // one 128B line per K row covers both 16-wide key tiles of the next block
            __builtin_prefetch(Kt_b + (size_t)lane * Tt + knext, 0, 0);
            __builtin_prefetch(Kt_b + (size_t)(32 + lane) * Tt + knext, 0, 0);
            __builtin_prefetch(V_b + (size_t)(knext + lane) * Hh, 0, 0);  // one row = one 128B line
        }

        // ---- S = (Q K^T): two 16x16 tiles covering key cols [kbase, kbase+32)
        v8f s0 = {}, s1 = {};
        s0 = WMMA_BF16(qf0, load_frag_b(Kt_b, Tt, 0,  kbase),      s0);
        s0 = WMMA_BF16(qf1, load_frag_b(Kt_b, Tt, 32, kbase),      s0);
        s1 = WMMA_BF16(qf0, load_frag_b(Kt_b, Tt, 0,  kbase + 16), s1);
        s1 = WMMA_BF16(qf1, load_frag_b(Kt_b, Tt, 32, kbase + 16), s1);

        const int ki0 = kbase + nlo;
        const int ki1 = ki0 + 16;
        const bool ok0 = am_b[ki0] != 0;
        const bool ok1 = am_b[ki1] != 0;

        float pr0[8], pr1[8], alpha[8];
#pragma unroll
        for (int r = 0; r < 8; ++r) {
            const int qi = qbase + r + 8 * hi;       // this VGPR's row
            float v0 = (ok0 && ki0 <= qi) ? s0[r] * scale : -1e30f;
            float v1 = (ok1 && ki1 <= qi) ? s1[r] * scale : -1e30f;
            // row max over the 32 key cols (reduction across a 16-lane half)
            float mx = fmaxf(v0, v1);
            mx = fmaxf(mx, __shfl_xor(mx, 1, 32));
            mx = fmaxf(mx, __shfl_xor(mx, 2, 32));
            mx = fmaxf(mx, __shfl_xor(mx, 4, 32));
            mx = fmaxf(mx, __shfl_xor(mx, 8, 32));
            const float mnew = fmaxf(mrow[r], mx);
            const float a  = __expf(mrow[r] - mnew);
            const float p0 = __expf(v0 - mnew);
            const float p1 = __expf(v1 - mnew);
            float ps = p0 + p1;
            ps += __shfl_xor(ps, 1, 32);
            ps += __shfl_xor(ps, 2, 32);
            ps += __shfl_xor(ps, 4, 32);
            ps += __shfl_xor(ps, 8, 32);
            lrow[r] = lrow[r] * a + ps;
            mrow[r] = mnew;
            alpha[r] = a;
            pr0[r] = p0;
            pr1[r] = p1;
        }

        // ---- reshape P (C-layout) -> A-layout via LDS; rescale O accumulators
#pragma unroll
        for (int r = 0; r < 8; ++r) {
            const int row = r + 8 * hi;
            Pls[row * 32 + nlo]      = f2bf(pr0[r]);
            Pls[row * 32 + nlo + 16] = f2bf(pr1[r]);
            o0[r] *= alpha[r];
            o1[r] *= alpha[r];
            o2[r] *= alpha[r];
            o3[r] *= alpha[r];
        }
        asm volatile("s_wait_dscnt 0" ::: "memory");   // same-wave cross-lane LDS RAW
        v16bf pf = load_frag_a(Pls, 32, 0, 0);

        // ---- O += P @ V  (4 N-tiles of 16 over H=64)
        o0 = WMMA_BF16(pf, load_frag_b(V_b, Hh, kbase, 0),  o0);
        o1 = WMMA_BF16(pf, load_frag_b(V_b, Hh, kbase, 16), o1);
        o2 = WMMA_BF16(pf, load_frag_b(V_b, Hh, kbase, 32), o2);
        o3 = WMMA_BF16(pf, load_frag_b(V_b, Hh, kbase, 48), o3);
    }

    // ---- normalize; padded query rows (mask==0) -> 0 (matches nan_to_num)
#pragma unroll
    for (int r = 0; r < 8; ++r) {
        const int qi = qbase + r + 8 * hi;
        const float inv = (am_b[qi] != 0 && lrow[r] > 0.f) ? 1.f / lrow[r] : 0.f;
        float* orow = out + ((size_t)b * Tt + qi) * Hh;
        orow[nlo]      = o0[r] * inv;
        orow[nlo + 16] = o1[r] * inv;
        orow[nlo + 32] = o2[r] * inv;
        orow[nlo + 48] = o3[r] * inv;
    }
}

extern "C" void kernel_launch(void* const* d_in, const int* in_sizes, int n_in,
                              void* d_out, int out_size, void* d_ws, size_t ws_size,
                              hipStream_t stream) {
    const float* x  = (const float*)d_in[0];
    const float* Wk = (const float*)d_in[1];
    const float* Wq = (const float*)d_in[2];
    const float* Wv = (const float*)d_in[3];
    const int*   am = (const int*)d_in[4];
    float* out = (float*)d_out;

    // Workspace layout (all 256B-aligned chunks; ~6.7 MB total)
    char* ws = (char*)d_ws;
    unsigned short* Wt  = (unsigned short*)ws; ws += (size_t)Ee * 192 * 2;  // 384 KB
    unsigned short* Qb  = (unsigned short*)ws; ws += (size_t)BT * Hh * 2;   // 2 MB
    unsigned short* Ktb = (unsigned short*)ws; ws += (size_t)BT * Hh * 2;   // 2 MB
    unsigned short* Vb  = (unsigned short*)ws; ws += (size_t)BT * Hh * 2;   // 2 MB

    cvt_w_kernel<<<(Ee * 192 + 255) / 256, 256, 0, stream>>>(Wk, Wq, Wv, Wt);
    proj_kernel<<<BT / 16, 128, 0, stream>>>(x, Wt, Qb, Ktb, Vb);
    attn_kernel<<<BT / 16, 32, 0, stream>>>(Qb, Ktb, Vb, am, out);
}
